// se_equivariant_net_29961691857583
// MI455X (gfx1250) — matrix operand order repaired and checked by reading, not compiled, not run on hardware
//
#include <hip/hip_runtime.h>
#include <cstdint>
#include <cstddef>

#define N_NODES 512
#define NBATCH  8
#define HDIM    32
#define UDIM    64
#define NLAYERS 3
#define NFEAT   32

typedef __attribute__((ext_vector_type(16))) __bf16 v16bf;
typedef __attribute__((ext_vector_type(8)))  float  v8f;

__device__ __forceinline__ float frcp(float x){ return __builtin_amdgcn_rcpf(x); }     // v_rcp_f32
__device__ __forceinline__ float fsqrt(float x){ return __builtin_amdgcn_sqrtf(x); }   // v_sqrt_f32
__device__ __forceinline__ float siluf(float x){ return x * frcp(1.0f + __expf(-x)); }
__device__ __forceinline__ float sigmf(float x){ return frcp(1.0f + __expf(-x)); }

__device__ __forceinline__ uint32_t f2bf(float x){
  uint32_t u = __float_as_uint(x);
  return (u + 0x7FFFu + ((u>>16)&1u)) >> 16;
}

union OpU { uint32_t d[8]; v16bf v; };

// A operand (16x32 bf16, row-major tile in LDS, Kt2 = cols/2 dwords per row)
__device__ __forceinline__ v16bf load_A(const uint32_t* lds, int Kt2, int s, int lane){
  int m = lane & 15, hi = lane >> 4;
  const uint32_t* p = lds + m*Kt2 + s*16 + hi*8;
  OpU u;
  *(uint4*)(&u.d[0]) = *(const uint4*)(p);
  *(uint4*)(&u.d[4]) = *(const uint4*)(p+4);
  return u.v;
}

// B operand from packed-B-layout LDS weight (K=32*S rows, 16*NT cols)
__device__ __forceinline__ v16bf load_Bpk(const uint32_t* pk, int S, int nt, int s, int lane){
  int n = lane & 15, hi = lane >> 4;
  const uint32_t* p = pk + ((((nt*S + s)<<1) + hi)<<7) + (n<<3);
  OpU u;
  *(uint4*)(&u.d[0]) = *(const uint4*)(p);
  *(uint4*)(&u.d[4]) = *(const uint4*)(p+4);
  return u.v;
}

// B operand with every column equal to vector w (wp = 32 dwords of bf16 pairs)
__device__ __forceinline__ v16bf load_Brep(const uint32_t* wp, int s, int lane){
  int hi = lane >> 4;
  const uint32_t* p = wp + s*16 + hi*8;
  OpU u;
  *(uint4*)(&u.d[0]) = *(const uint4*)(p);
  *(uint4*)(&u.d[4]) = *(const uint4*)(p+4);
  return u.v;
}

__device__ __forceinline__ v8f wmma_bf(v16bf a, v16bf b, v8f c){
  return __builtin_amdgcn_wmma_f32_16x16x32_bf16(false, a, false, b, (short)0, c, false, false);
}

// pack row-major f32 W[K][Ncols] (global) into B-operand-ordered bf16 dwords (LDS)
__device__ __forceinline__ void pack_B(const float* __restrict__ W, int K, int Ncols,
                                       uint32_t* dst, int tid, int nth){
  int S = K >> 5;
  int total = (Ncols >> 4) * S * 256;
  for (int idx = tid; idx < total; idx += nth){
    int v = idx & 7, n = (idx >> 3) & 15, hi = (idx >> 7) & 1, rest = idx >> 8;
    int s = rest % S, nt = rest / S;
    int k = 32*s + 16*hi + 2*v, c = 16*nt + n;
    dst[idx] = f2bf(W[k*Ncols + c]) | (f2bf(W[(k+1)*Ncols + c]) << 16);
  }
}

// stage f32 C-tile values (8 per lane) into row-major bf16 LDS tile (Cols wide)
__device__ __forceinline__ void stage_C(uint16_t* dst, int Cols, int nt, int lane, const float* vals){
  int c = (lane & 15) + (nt << 4), hi = lane >> 4;
  #pragma unroll
  for (int r = 0; r < 8; ++r) dst[(r + 8*hi)*Cols + c] = (uint16_t)f2bf(vals[r]);
}

__global__ __launch_bounds__(256)
void init_kernel(const float* __restrict__ x, float* __restrict__ xcur, float* __restrict__ hout){
  int idx = blockIdx.x*256 + threadIdx.x;
  if (idx < NBATCH*N_NODES*3)  xcur[idx] = x[idx];
  if (idx < NBATCH*N_NODES*HDIM) hout[idx] = 1.0f/(float)N_NODES;  // softmax+mean over same axis == 1/N
}

// per-node precompute: hjc = h@We0[1:33], hic = h@We0[33:65] + be0
__global__ __launch_bounds__(64)
void node_pre(const float* __restrict__ h, const float* __restrict__ We0, const float* __restrict__ be0,
              float* __restrict__ hjc, float* __restrict__ hic)
{
  const int node = blockIdx.x;
  const int t = threadIdx.x;
  __shared__ float hh[HDIM];
  if (t < HDIM) hh[t] = h[(size_t)node*HDIM + t];
  __syncthreads();
  float aj = 0.f, ai = be0[t];
  #pragma unroll 8
  for (int k = 0; k < HDIM; ++k){
    float hv = hh[k];
    aj += hv * We0[(1+k)*64 + t];
    ai += hv * We0[(33+k)*64 + t];
  }
  hjc[(size_t)node*64 + t] = aj;
  hic[(size_t)node*64 + t] = ai;
}

// fused pairwise kernel: one workgroup per (b, i); 4 waves over j-tiles of 16
__global__ __launch_bounds__(128)
void pair_kernel(const float* __restrict__ xcur,
                 const float* __restrict__ hjc, const float* __restrict__ hic,
                 const float* __restrict__ We0, const float* __restrict__ We1, const float* __restrict__ be1,
                 const float* __restrict__ Winf, const float* __restrict__ binf,
                 const float* __restrict__ Wx0, const float* __restrict__ bx0,
                 const float* __restrict__ Wx1, const float* __restrict__ bx1,
                 const float* __restrict__ Wxo, const float* __restrict__ bxo,
                 float* __restrict__ m_out, float* __restrict__ shift_out)
{
  __shared__ float lds_x[N_NODES*3];
  __shared__ float lds_wsq[64], lds_hic[64], lds_be1[64], lds_bx0[64], lds_bx1[64];
  __shared__ __align__(16) uint32_t pkWe1[2048], pkWx0[2048], pkWx1[2048];
  __shared__ __align__(16) uint32_t pkInf[32], pkXo[32];
  __shared__ float lds_mi[64];
  __shared__ float lds_sh[3];
  __shared__ __align__(16) uint16_t stA[4][16*64];
  __shared__ __align__(16) uint16_t stB[4][16*64];

  const int i = blockIdx.x, b = blockIdx.y;
  const int tid = threadIdx.x, lane = tid & 31, w = tid >> 5;
  const int node = b*N_NODES + i;

  for (int idx = tid; idx < N_NODES*3; idx += 128) lds_x[idx] = xcur[(size_t)b*N_NODES*3 + idx];
  if (tid < 64){
    lds_wsq[tid] = We0[tid];                       // row 0 of We0 (sqn weight)
    lds_hic[tid] = hic[(size_t)node*64 + tid];
    lds_be1[tid] = be1[tid]; lds_bx0[tid] = bx0[tid]; lds_bx1[tid] = bx1[tid];
    lds_mi[tid] = 0.f;
  }
  if (tid < 3) lds_sh[tid] = 0.f;
  pack_B(We1, 64, 64, pkWe1, tid, 128);
  pack_B(Wx0, 64, 64, pkWx0, tid, 128);
  pack_B(Wx1, 64, 64, pkWx1, tid, 128);
  for (int dI = tid; dI < 32; dI += 128){
    pkInf[dI] = f2bf(Winf[2*dI]) | (f2bf(Winf[2*dI+1]) << 16);
    pkXo[dI]  = f2bf(Wxo[2*dI])  | (f2bf(Wxo[2*dI+1])  << 16);
  }
  __syncthreads();

  const float xi0 = lds_x[i*3+0], xi1 = lds_x[i*3+1], xi2 = lds_x[i*3+2];
  const float binf_s = binf[0], bxo_s = bxo[0];

  uint16_t* sA = &stA[w][0]; const uint32_t* sA32 = (const uint32_t*)sA;
  uint16_t* sB = &stB[w][0]; const uint32_t* sB32 = (const uint32_t*)sB;

  float macc[4] = {0.f,0.f,0.f,0.f};
  float shacc = 0.f;
  const int cc = lane & 15, hi = lane >> 4;

  for (int jt = w; jt < 32; jt += 4){
    const int j0 = jt << 4;
    // build t0 = silu(sqn*wsq + hjc[j] + hic[i]) as bf16 A-tile (16x64)
    for (int it = 0; it < 32; ++it){
      int idx = it*32 + lane;
      int m = idx >> 6, k = idx & 63;
      int j = j0 + m;
      float dx = lds_x[j*3+0]-xi0, dy = lds_x[j*3+1]-xi1, dz = lds_x[j*3+2]-xi2;
      float sq = dx*dx + dy*dy + dz*dz;
      float v = sq*lds_wsq[k] + hjc[((size_t)b*N_NODES + j)*64 + k] + lds_hic[k];
      sA[m*64 + k] = (uint16_t)f2bf(siluf(v));
    }
    // phi_e layer 1: m_ij = silu(t0 @ We1 + be1)
    float msil[4][8];
    #pragma unroll
    for (int nt = 0; nt < 4; ++nt){
      float bias = lds_be1[(nt<<4) + cc];
      v8f c;
      #pragma unroll
      for (int r=0;r<8;++r) c[r]=bias;
      c = wmma_bf(load_A(sA32,32,0,lane), load_Bpk(pkWe1,2,nt,0,lane), c);
      c = wmma_bf(load_A(sA32,32,1,lane), load_Bpk(pkWe1,2,nt,1,lane), c);
      #pragma unroll
      for (int r=0;r<8;++r) msil[nt][r] = siluf(c[r]);
      stage_C(sB, 64, nt, lane, msil[nt]);
    }
    // e = sigmoid(m_ij @ Winf + binf) replicated across columns; zero diag
    float e_r[8];
    {
      v8f c;
      #pragma unroll
      for (int r=0;r<8;++r) c[r]=binf_s;
      c = wmma_bf(load_A(sB32,32,0,lane), load_Brep(pkInf,0,lane), c);
      c = wmma_bf(load_A(sB32,32,1,lane), load_Brep(pkInf,1,lane), c);
      #pragma unroll
      for (int r=0;r<8;++r){
        int j = j0 + r + 8*hi;
        e_r[r] = (j == i) ? 0.f : sigmf(c[r]);
      }
    }
    #pragma unroll
    for (int nt=0;nt<4;++nt){
      #pragma unroll
      for (int r=0;r<8;++r) macc[nt] += e_r[r]*msil[nt][r];
    }
    // phi_x layer 1: A=sB -> sA
    #pragma unroll
    for (int nt=0;nt<4;++nt){
      float bias = lds_bx0[(nt<<4)+cc];
      v8f c;
      #pragma unroll
      for (int r=0;r<8;++r) c[r]=bias;
      c = wmma_bf(load_A(sB32,32,0,lane), load_Bpk(pkWx0,2,nt,0,lane), c);
      c = wmma_bf(load_A(sB32,32,1,lane), load_Bpk(pkWx0,2,nt,1,lane), c);
      float vals[8];
      #pragma unroll
      for (int r=0;r<8;++r) vals[r]=siluf(c[r]);
      stage_C(sA, 64, nt, lane, vals);
    }
    // phi_x layer 2: A=sA -> sB
    #pragma unroll
    for (int nt=0;nt<4;++nt){
      float bias = lds_bx1[(nt<<4)+cc];
      v8f c;
      #pragma unroll
      for (int r=0;r<8;++r) c[r]=bias;
      c = wmma_bf(load_A(sA32,32,0,lane), load_Bpk(pkWx1,2,nt,0,lane), c);
      c = wmma_bf(load_A(sA32,32,1,lane), load_Bpk(pkWx1,2,nt,1,lane), c);
      float vals[8];
      #pragma unroll
      for (int r=0;r<8;++r) vals[r]=siluf(c[r]);
      stage_C(sB, 64, nt, lane, vals);
    }
    // px = u2 @ Wxo + bxo (replicated); shift accumulation (lanes cc<3, component cc)
    {
      v8f c;
      #pragma unroll
      for (int r=0;r<8;++r) c[r]=bxo_s;
      c = wmma_bf(load_A(sB32,32,0,lane), load_Brep(pkXo,0,lane), c);
      c = wmma_bf(load_A(sB32,32,1,lane), load_Brep(pkXo,1,lane), c);
      if (cc < 3){
        #pragma unroll
        for (int r=0;r<8;++r){
          int row = r + 8*hi; int j = j0 + row;
          if (j != i){
            float dx = lds_x[j*3+0]-xi0, dy = lds_x[j*3+1]-xi1, dz = lds_x[j*3+2]-xi2;
            float sq = dx*dx+dy*dy+dz*dz;
            float nr = fsqrt(sq + 1e-8f) + 1.0f;
            float dcc = lds_x[j*3+cc] - ((cc==0)?xi0:((cc==1)?xi1:xi2));
            shacc += c[r]*dcc*frcp(nr);
          }
        }
      }
    }
  }
  #pragma unroll
  for (int nt=0;nt<4;++nt) atomicAdd(&lds_mi[(nt<<4)+cc], macc[nt]);
  if (cc < 3) atomicAdd(&lds_sh[cc], shacc);
  __syncthreads();
  if (tid < 64) m_out[(size_t)node*64 + tid] = lds_mi[tid];
  if (tid < 3)  shift_out[node*3 + tid] = lds_sh[tid];
}

// per-node update: x += shift/(N-1);  h += phi_h_out(phi_h_mlp([m_i, h]))
__global__ __launch_bounds__(64)
void node_update(float* __restrict__ xcur, float* __restrict__ h,
                 const float* __restrict__ m_i, const float* __restrict__ shift,
                 const float* __restrict__ Wh0, const float* __restrict__ bh0,
                 const float* __restrict__ Wh1, const float* __restrict__ bh1,
                 const float* __restrict__ Who, const float* __restrict__ bho)
{
  const int node = blockIdx.x;
  const int t = threadIdx.x;
  __shared__ float in96[96];
  __shared__ float u1[64];
  __shared__ float u2[64];
  in96[t] = m_i[(size_t)node*64 + t];
  if (t < 32) in96[64+t] = h[(size_t)node*HDIM + t];
  __syncthreads();
  float a = bh0[t];
  for (int k=0;k<96;++k) a += in96[k]*Wh0[k*64+t];
  u1[t] = siluf(a);
  __syncthreads();
  float a2 = bh1[t];
  for (int k=0;k<64;++k) a2 += u1[k]*Wh1[k*64+t];
  u2[t] = a2;  // activate_final=False
  __syncthreads();
  if (t < 32){
    float d = bho[t];
    for (int k=0;k<64;++k) d += u2[k]*Who[k*32+t];
    h[(size_t)node*HDIM + t] += d;
  }
  if (t < 3) xcur[node*3 + t] += shift[node*3 + t] * (1.0f/(float)(N_NODES-1));
}

// final: softmax(sq0-embed) over j, final_mlp via WMMA, mean over j, concat softmax(h), h_final
__global__ __launch_bounds__(128)
void final_kernel(const float* __restrict__ xorig, const float* __restrict__ xcur,
                  const float* __restrict__ h,
                  const float* __restrict__ Ws, const float* __restrict__ bs,
                  const float* __restrict__ Wf0, const float* __restrict__ bf0,
                  const float* __restrict__ Wf1, const float* __restrict__ bf1,
                  const float* __restrict__ Wf2, const float* __restrict__ bf2,
                  const float* __restrict__ Whf, const float* __restrict__ bhf,
                  float* __restrict__ out_x, float* __restrict__ out_h)
{
  __shared__ float lds_x[N_NODES*3];
  __shared__ float lds_sq[N_NODES];
  __shared__ __align__(16) uint32_t pkW0[1024], pkW1[2048], pkW2[1024];
  __shared__ float sums[32];
  __shared__ float rsums[32];
  __shared__ float maccL[32];
  __shared__ __align__(16) uint16_t stg [4][16*64];
  __shared__ __align__(16) uint16_t stg2[4][16*64];
  __shared__ float catv[64];
  __shared__ float htmp[32];
  __shared__ float red2[2];
  __shared__ int maxI, minI;

  const int i = blockIdx.x, b = blockIdx.y;
  const int tid = threadIdx.x, lane = tid & 31, w = tid >> 5;
  const int node = b*N_NODES + i;

  for (int idx = tid; idx < N_NODES*3; idx += 128) lds_x[idx] = xorig[(size_t)b*N_NODES*3 + idx];
  if (tid < 32){ sums[tid] = 0.f; maccL[tid] = 0.f; }
  if (tid == 0){ maxI = 0; minI = 0x7F7FFFFF; }
  pack_B(Wf0, 32, 64, pkW0, tid, 128);
  pack_B(Wf1, 64, 64, pkW1, tid, 128);
  pack_B(Wf2, 64, 32, pkW2, tid, 128);
  __syncthreads();

  const float xi0 = lds_x[i*3], xi1 = lds_x[i*3+1], xi2 = lds_x[i*3+2];
  for (int j = tid; j < N_NODES; j += 128){
    if (j == i){ lds_sq[j] = 0.f; continue; }
    float dx = lds_x[j*3]-xi0, dy = lds_x[j*3+1]-xi1, dz = lds_x[j*3+2]-xi2;
    float sq = dx*dx+dy*dy+dz*dz;
    lds_sq[j] = sq;
    atomicMax(&maxI, __float_as_int(sq));   // sq >= 0: int order == float order
    atomicMin(&minI, __float_as_int(sq));
  }
  __syncthreads();
  const float sqmax = __int_as_float(maxI), sqmin = __int_as_float(minI);
  {
    int c = tid & 31, g = tid >> 5;
    float wc = Ws[c], bc = bs[c];
    float Ml = fmaxf(sqmax*wc, sqmin*wc) + bc;
    float loc = 0.f;
    for (int j = g; j < N_NODES; j += 4)
      if (j != i) loc += __expf(lds_sq[j]*wc + bc - Ml);
    atomicAdd(&sums[c], loc);
  }
  __syncthreads();
  if (tid < 32) rsums[tid] = frcp(sums[tid]);
  __syncthreads();

  uint16_t* s1 = &stg[w][0];  const uint32_t* s1_32 = (const uint32_t*)s1;
  uint16_t* s2 = &stg2[w][0]; const uint32_t* s2_32 = (const uint32_t*)s2;
  float acc[2] = {0.f, 0.f};
  const int cc = lane & 15, hi = lane >> 4;

  for (int jt = w; jt < 32; jt += 4){
    int j0 = jt << 4;
    for (int it = 0; it < 16; ++it){            // s-tile 16x32 bf16
      int idx = it*32 + lane;
      int m = idx >> 5, c = idx & 31;
      int j = j0 + m;
      float val = 0.f;
      if (j != i){
        float wc = Ws[c], bc = bs[c];
        float Ml = fmaxf(sqmax*wc, sqmin*wc) + bc;
        val = __expf(lds_sq[j]*wc + bc - Ml) * rsums[c];
      }
      s1[m*32 + c] = (uint16_t)f2bf(val);
    }
    #pragma unroll
    for (int nt=0;nt<4;++nt){                   // L0: 32 -> 64
      float bias = bf0[(nt<<4)+cc];
      v8f c;
      #pragma unroll
      for (int r=0;r<8;++r) c[r]=bias;
      c = wmma_bf(load_A(s1_32,16,0,lane), load_Bpk(pkW0,1,nt,0,lane), c);
      float vals[8];
      #pragma unroll
      for (int r=0;r<8;++r) vals[r]=siluf(c[r]);
      stage_C(s2, 64, nt, lane, vals);
    }
    #pragma unroll
    for (int nt=0;nt<4;++nt){                   // L1: 64 -> 64
      float bias = bf1[(nt<<4)+cc];
      v8f c;
      #pragma unroll
      for (int r=0;r<8;++r) c[r]=bias;
      c = wmma_bf(load_A(s2_32,32,0,lane), load_Bpk(pkW1,2,nt,0,lane), c);
      c = wmma_bf(load_A(s2_32,32,1,lane), load_Bpk(pkW1,2,nt,1,lane), c);
      float vals[8];
      #pragma unroll
      for (int r=0;r<8;++r) vals[r]=siluf(c[r]);
      stage_C(s1, 64, nt, lane, vals);
    }
    #pragma unroll
    for (int nt=0;nt<2;++nt){                   // L2: 64 -> 32, accumulate row-sums
      float bias = bf2[(nt<<4)+cc];
      v8f c;
      #pragma unroll
      for (int r=0;r<8;++r) c[r]=bias;
      c = wmma_bf(load_A(s1_32,32,0,lane), load_Bpk(pkW2,2,nt,0,lane), c);
      c = wmma_bf(load_A(s1_32,32,1,lane), load_Bpk(pkW2,2,nt,1,lane), c);
      float srow = 0.f;
      #pragma unroll
      for (int r=0;r<8;++r) srow += siluf(c[r]);
      acc[nt] += srow;
    }
  }
  atomicAdd(&maccL[cc],      acc[0]);
  atomicAdd(&maccL[16+cc],   acc[1]);
  if (tid < 32) htmp[tid] = h[(size_t)node*HDIM + tid];
  __syncthreads();
  if (tid == 0){
    float mx = -1e30f;
    for (int k=0;k<32;++k) mx = fmaxf(mx, htmp[k]);
    float sm = 0.f;
    for (int k=0;k<32;++k) sm += __expf(htmp[k]-mx);
    red2[0]=mx; red2[1]=frcp(sm);
  }
  __syncthreads();
  if (tid < 32){
    catv[tid]    = maccL[tid] * (1.0f/(float)N_NODES);
    catv[32+tid] = __expf(htmp[tid]-red2[0])*red2[1];
  }
  __syncthreads();
  if (tid < 32){
    float o = bhf[tid];
    for (int k=0;k<64;++k) o += catv[k]*Whf[k*32+tid];
    out_h[(size_t)node*32 + tid] = o;
  }
  if (tid < 3) out_x[node*3 + tid] = xcur[node*3 + tid];
}

extern "C" void kernel_launch(void* const* d_in, const int* in_sizes, int n_in,
                              void* d_out, int out_size, void* d_ws, size_t ws_size,
                              hipStream_t stream)
{
  (void)out_size; (void)ws_size;
  const float* const* in = (const float* const*)d_in;
  const float* X; int base;
  if (n_in > 0 && in_sizes[0] == NBATCH*N_NODES*3){ X = in[0]; base = 1; }   // x first
  else { X = in[n_in-1]; base = 0; }                                         // x last (sorted keys)

  // params flattened (sorted keys): final_mlp(6), h_embed(2, unused), h_final(2), layers(3*18), sq_embed(2)
  const float* Wf0  = in[base+0]; const float* bf0v = in[base+1];
  const float* Wf1  = in[base+2]; const float* bf1v = in[base+3];
  const float* Wf2  = in[base+4]; const float* bf2v = in[base+5];
  const float* Whf  = in[base+8]; const float* bhfv = in[base+9];
  const float* Wsq  = in[base+64]; const float* bsq = in[base+65];

  float* ws   = (float*)d_ws;
  float* xcur = ws;
  float* h    = xcur + NBATCH*N_NODES*3;
  float* hjc  = h    + NBATCH*N_NODES*HDIM;
  float* hic  = hjc  + NBATCH*N_NODES*64;
  float* mi   = hic  + NBATCH*N_NODES*64;
  float* shf  = mi   + NBATCH*N_NODES*64;

  init_kernel<<<512, 256, 0, stream>>>(X, xcur, h);
  for (int l = 0; l < NLAYERS; ++l){
    // per-layer (sorted): 0 We0,1 be0,2 We1,3 be1, 4 Wh0,5 bh0,6 Wh1,7 bh1, 8 Who,9 bho,
    //                     10 Winf,11 binf, 12 Wx0,13 bx0,14 Wx1,15 bx1, 16 Wxo,17 bxo
    const float* const* L = in + base + 10 + 18*l;
    node_pre<<<NBATCH*N_NODES, 64, 0, stream>>>(h, L[0], L[1], hjc, hic);
    pair_kernel<<<dim3(N_NODES, NBATCH), 128, 0, stream>>>(
        xcur, hjc, hic,
        L[0], L[2], L[3], L[10], L[11], L[12], L[13], L[14], L[15], L[16], L[17],
        mi, shf);
    node_update<<<NBATCH*N_NODES, 64, 0, stream>>>(xcur, h, mi, shf,
        L[4], L[5], L[6], L[7], L[8], L[9]);
  }
  float* out_x = (float*)d_out;
  float* out_h = out_x + NBATCH*N_NODES*3;
  final_kernel<<<dim3(N_NODES, NBATCH), 128, 0, stream>>>(
      X, xcur, h, Wsq, bsq, Wf0, bf0v, Wf1, bf1v, Wf2, bf2v, Whf, bhfv, out_x, out_h);
}
